// UnifiedMemory_26680336843535
// MI455X (gfx1250) — matrix operand order repaired and checked by reading, not compiled, not run on hardware
//
#include <hip/hip_runtime.h>
#include <hip/hip_bf16.h>

typedef __attribute__((ext_vector_type(16))) __bf16 v16bf;
typedef __attribute__((ext_vector_type(8)))  float  v8f;
typedef __attribute__((ext_vector_type(4)))  float  v4f;

#define B_ROWS 64
#define N_FEAT 65536
#define D_DIM  256
#define NQ     25

// ---------------------------------------------------------------------------
// Kernel 0: zero the nbs/rec accumulators in workspace (3200 floats)
// ---------------------------------------------------------------------------
__global__ void zero_ws_kernel(float* ws, int n) {
    for (int i = threadIdx.x; i < n; i += blockDim.x) ws[i] = 0.0f;
}

// ---------------------------------------------------------------------------
// Kernel 1: fused GEMM (bf16 WMMA) + tent-quantizer bin accumulation
//           + feature copy to output (saves a second 64 MB read of features).
// Grid: 512 blocks x 256 threads (8 waves). Each wave owns 16 columns of N.
// Block covers 128 columns; 512*128 = 65536 = N.
// ---------------------------------------------------------------------------
__global__ __launch_bounds__(256)
void gemm_quant_kernel(const float* __restrict__ inputs,    // [64][256]
                       const float* __restrict__ features,  // [65536][256]
                       const int*   __restrict__ labels,    // [65536]
                       const int*   __restrict__ indexes,   // [64]
                       float* __restrict__ ws_nbs,          // [64*25]
                       float* __restrict__ ws_rec,          // [64*25]
                       float* __restrict__ feat_out) {      // [65536][256] (4B aligned only)
    // A-operand staging: 4 M-tiles x 8 K-steps x 32 lanes x 16 bf16 = 32 KB
    __shared__ __align__(32) __bf16 lds_a[4 * 8 * 32 * 16];
    __shared__ float nbs_l[B_ROWS * NQ];
    __shared__ float rec_l[B_ROWS * NQ];
    __shared__ int   lq[B_ROWS];

    const int tid  = threadIdx.x;
    const int lane = tid & 31;
    const int wave = tid >> 5;

    // ---- stage inputs into WMMA A-matrix bf16 layout --------------------
    // lane 0-15  : M = lane,    K = {0..15} (+32*ks)
    // lane 16-31 : M = lane-16, K = {16..31} (+32*ks)
    for (int s = tid; s < 4 * 8 * 32; s += 256) {
        const int mt = s >> 8;
        const int ks = (s >> 5) & 7;
        const int ln = s & 31;
        const int m  = mt * 16 + (ln & 15);
        const int kb = ks * 32 + ((ln < 16) ? 0 : 16);
        const float* ir = inputs + m * D_DIM;
        __bf16* dst = &lds_a[s * 16];
#pragma unroll
        for (int j = 0; j < 8; ++j) dst[j]     = (__bf16)ir[kb + j];
#pragma unroll
        for (int j = 0; j < 8; ++j) dst[8 + j] = (__bf16)ir[kb + 8 + j];
    }
    if (tid < B_ROWS) lq[tid] = labels[indexes[tid]];
    for (int i = tid; i < B_ROWS * NQ; i += 256) { nbs_l[i] = 0.0f; rec_l[i] = 0.0f; }
    __syncthreads();

    // ---- GEMM: this wave's 16 columns, all 64 rows ----------------------
    const int n0 = blockIdx.x * 128 + wave * 16;
    const int n  = n0 + (lane & 15);

    v8f acc[4];
#pragma unroll
    for (int mt = 0; mt < 4; ++mt)
#pragma unroll
        for (int r = 0; r < 8; ++r) acc[mt][r] = 0.0f;

    const size_t rowoff = (size_t)n * D_DIM + ((lane >> 4) << 4);
    const float* frow_base = features + rowoff;
    float*       drow_base = feat_out + rowoff;   // fused copy destination
#pragma unroll
    for (int ks = 0; ks < 8; ++ks) {
        // B-operand: lane = column n, 16 contiguous K values (fp32 -> bf16)
        const float* frow = frow_base + ks * 32;
        v4f f0 = *(const v4f*)(frow + 0);
        v4f f1 = *(const v4f*)(frow + 4);
        v4f f2 = *(const v4f*)(frow + 8);
        v4f f3 = *(const v4f*)(frow + 12);

        // fused feature copy: destination is only 4B aligned (out+1), so
        // dword stores; each lane writes its own contiguous 64B run.
        float* drow = drow_base + ks * 32;
#pragma unroll
        for (int j = 0; j < 4; ++j) {
            drow[j]      = f0[j];
            drow[4 + j]  = f1[j];
            drow[8 + j]  = f2[j];
            drow[12 + j] = f3[j];
        }

        v16bf bmat;
#pragma unroll
        for (int j = 0; j < 4; ++j) {
            bmat[j]      = (__bf16)f0[j];
            bmat[4 + j]  = (__bf16)f1[j];
            bmat[8 + j]  = (__bf16)f2[j];
            bmat[12 + j] = (__bf16)f3[j];
        }
#pragma unroll
        for (int mt = 0; mt < 4; ++mt) {
            v16bf amat = *(const v16bf*)&lds_a[((mt * 8 + ks) * 32 + lane) * 16];
            acc[mt] = __builtin_amdgcn_wmma_f32_16x16x32_bf16(
                false, amat, false, bmat, (short)0, acc[mt], false, false);
        }
    }

    // ---- tent quantizer: each sim value hits exactly 2 adjacent bins ----
    const int myl = labels[n];
#pragma unroll
    for (int mt = 0; mt < 4; ++mt) {
#pragma unroll
        for (int r = 0; r < 8; ++r) {
            const float s = acc[mt][r];
            const int   m = mt * 16 + ((lane >> 4) << 3) + r;
            float t = (1.0f - s) * 12.0f;          // bin-space position, 0..24
            t = fminf(fmaxf(t, 0.0f), 24.0f);
            int k0 = (int)t;
            if (k0 > 23) k0 = 23;
            const float f  = t - (float)k0;
            const float w0 = 1.0f - f;
            atomicAdd(&nbs_l[m * NQ + k0],     w0);
            atomicAdd(&nbs_l[m * NQ + k0 + 1], f);
            if (myl == lq[m]) {
                atomicAdd(&rec_l[m * NQ + k0],     w0);
                atomicAdd(&rec_l[m * NQ + k0 + 1], f);
            }
        }
    }
    __syncthreads();

    // ---- flush block partials to global accumulators --------------------
    for (int i = tid; i < B_ROWS * NQ; i += 256) {
        atomicAdd(ws_nbs + i, nbs_l[i]);
        atomicAdd(ws_rec + i, rec_l[i]);
    }
}

// ---------------------------------------------------------------------------
// Kernel 2: AP loss finalization. One block, 64 threads (one row each).
// ---------------------------------------------------------------------------
__global__ void finalize_loss_kernel(const float* __restrict__ ws_nbs,
                                     const float* __restrict__ ws_rec,
                                     float* __restrict__ out) {
    __shared__ float ap_s[B_ROWS];
    const int m = threadIdx.x;
    const float* nb = ws_nbs + m * NQ;
    const float* rc = ws_rec + m * NQ;

    float tot = 0.0f;
    for (int k = 0; k < NQ; ++k) tot += rc[k];

    float cn = 0.0f, cr = 0.0f, ap = 0.0f;
    for (int k = 0; k < NQ; ++k) {
        cn += nb[k];
        cr += rc[k];
        const float prec = cr / (1e-16f + cn);
        ap += prec * (rc[k] / tot);
    }
    ap_s[m] = ap;
    __syncthreads();
    if (m == 0) {
        float sum = 0.0f;
        for (int i = 0; i < B_ROWS; ++i) sum += ap_s[i];
        out[0] = 1.0f - sum / (float)B_ROWS;
    }
}

// ---------------------------------------------------------------------------
// Kernel 3: sequential momentum update (order matters for duplicate indexes).
// One block of 256 threads (= D); LDS tree reduction for the row norm.
// Runs after gemm_quant_kernel, which produced the feature copy in feat_out.
// ---------------------------------------------------------------------------
__global__ void seq_update_kernel(const float* __restrict__ inputs,
                                  const int*   __restrict__ indexes,
                                  float* __restrict__ feat_out) {
    __shared__ float red[D_DIM];
    const int tid = threadIdx.x;
    for (int b = 0; b < B_ROWS; ++b) {
        const int y = indexes[b];
        float v = 0.2f * feat_out[(size_t)y * D_DIM + tid]
                + 0.8f * inputs[b * D_DIM + tid];
        red[tid] = v * v;
        __syncthreads();
        for (int off = 128; off > 0; off >>= 1) {
            if (tid < off) red[tid] += red[tid + off];
            __syncthreads();
        }
        const float inv = 1.0f / sqrtf(red[0]);
        feat_out[(size_t)y * D_DIM + tid] = v * inv;
        __syncthreads();
    }
}

// ---------------------------------------------------------------------------
extern "C" void kernel_launch(void* const* d_in, const int* in_sizes, int n_in,
                              void* d_out, int out_size, void* d_ws, size_t ws_size,
                              hipStream_t stream) {
    const float* inputs   = (const float*)d_in[0];  // [64*256]
    const float* features = (const float*)d_in[1];  // [65536*256]
    const int*   labels   = (const int*)d_in[2];    // [65536]
    const int*   indexes  = (const int*)d_in[3];    // [64]

    float* out    = (float*)d_out;          // [0] = loss, [1..] = new_features
    float* ws_nbs = (float*)d_ws;           // 64*25
    float* ws_rec = ws_nbs + B_ROWS * NQ;   // 64*25

    zero_ws_kernel<<<1, 256, 0, stream>>>(ws_nbs, 2 * B_ROWS * NQ);

    gemm_quant_kernel<<<N_FEAT / 128, 256, 0, stream>>>(
        inputs, features, labels, indexes, ws_nbs, ws_rec, out + 1);

    finalize_loss_kernel<<<1, B_ROWS, 0, stream>>>(ws_nbs, ws_rec, out);

    seq_update_kernel<<<1, D_DIM, 0, stream>>>(inputs, indexes, out + 1);
}
